// GroupQueryAttention_46514495815691
// MI455X (gfx1250) — matrix operand order repaired and checked by reading, not compiled, not run on hardware
//
#include <hip/hip_runtime.h>
#include <math.h>

typedef __attribute__((ext_vector_type(16))) __bf16 bf16x16;
typedef __attribute__((ext_vector_type(8)))  __bf16 bf16x8;
typedef __attribute__((ext_vector_type(8)))  float  f32x8;

namespace {
constexpr int kDim  = 1024;
constexpr int kSeq  = 2048;
constexpr int kBs   = 2;
constexpr int kTok  = kBs * kSeq;      // 4096 tokens
constexpr int kGrp  = 8;
constexpr int kHpg  = 4;               // heads per group
constexpr int kHd   = 32;              // head dim
constexpr int kKvd  = kGrp * kHd;      // 256
constexpr int kNcol = kDim + 2 * kKvd; // 1536 fused output cols (q|k|v)
constexpr float kLog2e = 1.44269504088896340736f;
}

__device__ __forceinline__ f32x8 wmma_bf16(bf16x16 a, bf16x16 b, f32x8 c) {
  return __builtin_amdgcn_wmma_f32_16x16x32_bf16(false, a, false, b, (short)0, c,
                                                 false, false);
}

__device__ __forceinline__ bf16x16 cat8(bf16x8 lo, bf16x8 hi) {
  return __builtin_shufflevector(lo, hi, 0, 1, 2, 3, 4, 5, 6, 7, 8, 9, 10, 11,
                                 12, 13, 14, 15);
}

__device__ __forceinline__ float half16_max(float v) {
  // reduce across the 16 lanes of this lane-half (xor masks stay within half)
  v = fmaxf(v, __shfl_xor(v, 1));
  v = fmaxf(v, __shfl_xor(v, 2));
  v = fmaxf(v, __shfl_xor(v, 4));
  v = fmaxf(v, __shfl_xor(v, 8));
  return v;
}
__device__ __forceinline__ float half16_sum(float v) {
  v += __shfl_xor(v, 1);
  v += __shfl_xor(v, 2);
  v += __shfl_xor(v, 4);
  v += __shfl_xor(v, 8);
  return v;
}

// ---------------------------------------------------------------- fp32 -> bf16
// vectorized: 8 floats (b128 x2) -> 8 bf16 (b128) per thread per step
__global__ void cvt_bf16_kernel(const float* __restrict__ src,
                                __bf16* __restrict__ dst, int n8) {
  int i = blockIdx.x * blockDim.x + threadIdx.x;
  int stride = gridDim.x * blockDim.x;
  for (; i < n8; i += stride) {
    f32x8 f = *(const f32x8*)(src + (size_t)i * 8);
    bf16x8 o;
#pragma unroll
    for (int j = 0; j < 8; ++j) o[j] = (__bf16)f[j];
    *(bf16x8*)(dst + (size_t)i * 8) = o;
  }
}

// ------------------------------------------------- fused QKV projection GEMM
// C[tok, col] = X[tok, :] . W[col, :]   (torch Linear: x @ W.T)
// cols 0..1023 -> q (natural layout), 1024..1279 -> k (natural),
// 1280..1535 -> v written TRANSPOSED as [b][feature][seq].
// Register blocking: each wave computes 2 row-tiles x 4 col-tiles (32x64).
__global__ __launch_bounds__(256) void qkv_proj_kernel(
    const __bf16* __restrict__ xb, const __bf16* __restrict__ wb,
    __bf16* __restrict__ qb, __bf16* __restrict__ kb,
    __bf16* __restrict__ vtb) {
  const int lane = threadIdx.x & 31;
  const int wave = threadIdx.x >> 5;
  const int h    = lane >> 4;       // lane half (A: K-offset select)
  const int l16  = lane & 15;
  const int wr   = wave >> 2;       // 0..1  (row sub-block)
  const int wc   = wave & 3;        // 0..3  (col sub-block)
  const int row0 = blockIdx.x * 64 + wr * 32;   // 2 row tiles: row0, row0+16
  const int cbase = blockIdx.y * 256 + wc * 64; // 4 col tiles of 16

  const __bf16* arow0 = xb + (size_t)(row0 + l16) * kDim;
  const __bf16* arow1 = xb + (size_t)(row0 + 16 + l16) * kDim;
  const __bf16* brow0 = wb + (size_t)(cbase + l16) * kDim;
  const __bf16* brow1 = wb + (size_t)(cbase + 16 + l16) * kDim;
  const __bf16* brow2 = wb + (size_t)(cbase + 32 + l16) * kDim;
  const __bf16* brow3 = wb + (size_t)(cbase + 48 + l16) * kDim;

  f32x8 acc[2][4] = {};
  for (int kk = 0; kk < kDim; kk += 32) {
    // A 16x32 bf16, k order chosen here (same as B): lane holds row M=lane%16, K = i + 16*(lane/16)
    bf16x16 a0 = *(const bf16x16*)(arow0 + kk + 16 * h);
    bf16x16 a1 = *(const bf16x16*)(arow1 + kk + 16 * h);
    // B 32x16 bf16, k order chosen here (same as A): lane holds column N=lane%16, K = i + 16*(lane/16)
    bf16x16 b0 = *(const bf16x16*)(brow0 + kk + 16 * h);
    bf16x16 b1 = *(const bf16x16*)(brow1 + kk + 16 * h);
    bf16x16 b2 = *(const bf16x16*)(brow2 + kk + 16 * h);
    bf16x16 b3 = *(const bf16x16*)(brow3 + kk + 16 * h);
    acc[0][0] = wmma_bf16(a0, b0, acc[0][0]);
    acc[0][1] = wmma_bf16(a0, b1, acc[0][1]);
    acc[0][2] = wmma_bf16(a0, b2, acc[0][2]);
    acc[0][3] = wmma_bf16(a0, b3, acc[0][3]);
    acc[1][0] = wmma_bf16(a1, b0, acc[1][0]);
    acc[1][1] = wmma_bf16(a1, b1, acc[1][1]);
    acc[1][2] = wmma_bf16(a1, b2, acc[1][2]);
    acc[1][3] = wmma_bf16(a1, b3, acc[1][3]);
  }

  // D layout: VGPR r, lane -> (M = r + 8*(lane/16), N = lane%16)
#pragma unroll
  for (int rt = 0; rt < 2; ++rt) {
#pragma unroll
    for (int ci = 0; ci < 4; ++ci) {
      const int col = cbase + ci * 16 + l16;
#pragma unroll
      for (int r = 0; r < 8; ++r) {
        const int t = row0 + rt * 16 + r + 8 * h;  // global token index
        const __bf16 v = (__bf16)acc[rt][ci][r];
        if (col < kDim) {
          qb[(size_t)t * kDim + col] = v;
        } else if (col < kDim + kKvd) {
          kb[(size_t)t * kKvd + (col - kDim)] = v;
        } else {
          const int vc = col - (kDim + kKvd);  // feature = g*32 + d
          const int bidx = t >> 11;            // t / 2048
          const int s    = t & 2047;
          vtb[((size_t)(bidx * kKvd + vc)) * kSeq + s] = v;
        }
      }
    }
  }
}

// ---------------------------------------------------- flash-style GQA kernel
// block = 4 waves (one per head in group), grid = (seq/16, groups, batch)
// 64 keys per iteration: 4 S-WMMAs + 4 PV-WMMAs; deferred row-sum reduction.
__global__ __launch_bounds__(128) void gqa_attn_kernel(
    const __bf16* __restrict__ qb, const __bf16* __restrict__ kb,
    const __bf16* __restrict__ vtb, float* __restrict__ out) {
  const int lane = threadIdx.x & 31;
  const int m    = threadIdx.x >> 5;   // head within group (wave id)
  const int h    = lane >> 4;
  const int l16  = lane & 15;
  const int qt   = blockIdx.x;         // 16-query tile
  const int g    = blockIdx.y;
  const int b    = blockIdx.z;
  const int ho   = (g * kHpg + m) * kHd;  // feature offset of this head

  // per-wave P transpose buffer; 72-elem row stride breaks bank pattern
  __shared__ __align__(16) __bf16 pbuf[4][16][72];

  // Q A-operand (reused across all key tiles)
  const __bf16* qrow = qb + (size_t)(b * kSeq + qt * 16 + l16) * kDim + ho;
  bf16x16 aq = *(const bf16x16*)(qrow + 16 * h);

  float rowMax[8], sumP[8];
  f32x8 o0 = {}, o1 = {};
#pragma unroll
  for (int r = 0; r < 8; ++r) { rowMax[r] = -INFINITY; sumP[r] = 0.0f; }

  // K: [b][key][256]; B column n = key row (contiguous in d)
  const __bf16* kbase = kb + (size_t)b * kSeq * kKvd + g * kHd + 16 * h;
  // V transposed: [b][feature][seq]; B column n = feature, contiguous keys
  const __bf16* v0base =
      vtb + ((size_t)(b * kKvd + g * kHd) + l16) * kSeq + 16 * h;
  const __bf16* v1base =
      vtb + ((size_t)(b * kKvd + g * kHd + 16) + l16) * kSeq + 16 * h;

  for (int k0 = 0; k0 < kSeq; k0 += 64) {
    // ---- S = Q . K^T for 64 keys (four 16-key tiles)
    bf16x16 bk0 = *(const bf16x16*)(kbase + (size_t)(k0 + l16) * kKvd);
    bf16x16 bk1 = *(const bf16x16*)(kbase + (size_t)(k0 + 16 + l16) * kKvd);
    bf16x16 bk2 = *(const bf16x16*)(kbase + (size_t)(k0 + 32 + l16) * kKvd);
    bf16x16 bk3 = *(const bf16x16*)(kbase + (size_t)(k0 + 48 + l16) * kKvd);
    __builtin_prefetch(kbase + (size_t)(k0 + 64 + l16) * kKvd, 0, 1);
    f32x8 z = {};
    f32x8 s0 = wmma_bf16(aq, bk0, z);
    f32x8 s1 = wmma_bf16(aq, bk1, z);
    f32x8 s2 = wmma_bf16(aq, bk2, z);
    f32x8 s3 = wmma_bf16(aq, bk3, z);

    // ---- online softmax (row = r + 8*h spans the 16 lanes of this half)
#pragma unroll
    for (int r = 0; r < 8; ++r) {
      float tmax = fmaxf(fmaxf(s0[r], s1[r]), fmaxf(s2[r], s3[r]));
      tmax = half16_max(tmax);
      float nm = fmaxf(rowMax[r], tmax);
      float al = __builtin_exp2f((rowMax[r] - nm) * kLog2e);
      rowMax[r] = nm;
      float p0 = __builtin_exp2f((s0[r] - nm) * kLog2e);
      float p1 = __builtin_exp2f((s1[r] - nm) * kLog2e);
      float p2 = __builtin_exp2f((s2[r] - nm) * kLog2e);
      float p3 = __builtin_exp2f((s3[r] - nm) * kLog2e);
      // deferred reduction: per-lane partial sum (alpha is row-uniform)
      sumP[r] = sumP[r] * al + ((p0 + p1) + (p2 + p3));
      o0[r] *= al;
      o1[r] *= al;
      // stage P (D layout) into LDS row-major [q][key]
      pbuf[m][r + 8 * h][l16]      = (__bf16)p0;
      pbuf[m][r + 8 * h][16 + l16] = (__bf16)p1;
      pbuf[m][r + 8 * h][32 + l16] = (__bf16)p2;
      pbuf[m][r + 8 * h][48 + l16] = (__bf16)p3;
    }

    // same-wave LDS RAW: make stores visible before A-layout reload
    asm volatile("s_wait_dscnt 0" ::: "memory");

    // reload P in A-matrix layout (16 queries x 2x32 keys)
    bf16x16 ap0 = cat8(*(const bf16x8*)&pbuf[m][l16][16 * h],
                       *(const bf16x8*)&pbuf[m][l16][8 + 16 * h]);
    bf16x16 ap1 = cat8(*(const bf16x8*)&pbuf[m][l16][32 + 16 * h],
                       *(const bf16x8*)&pbuf[m][l16][40 + 16 * h]);

    // ---- O += P . V (two d-tiles x two 32-key chunks)
    bf16x16 bv00 = *(const bf16x16*)(v0base + k0);
    bf16x16 bv01 = *(const bf16x16*)(v0base + k0 + 32);
    bf16x16 bv10 = *(const bf16x16*)(v1base + k0);
    bf16x16 bv11 = *(const bf16x16*)(v1base + k0 + 32);
    __builtin_prefetch(v0base + k0 + 64, 0, 1);
    o0 = wmma_bf16(ap0, bv00, o0);
    o0 = wmma_bf16(ap1, bv01, o0);
    o1 = wmma_bf16(ap0, bv10, o1);
    o1 = wmma_bf16(ap1, bv11, o1);
  }

  // ---- final cross-lane sum reduction, normalize, store fp32 output
  float* obase = out + (size_t)(b * kSeq + qt * 16) * kDim + ho;
#pragma unroll
  for (int r = 0; r < 8; ++r) {
    const float inv = 1.0f / half16_sum(sumP[r]);
    obase[(size_t)(r + 8 * h) * kDim + l16]      = o0[r] * inv;
    obase[(size_t)(r + 8 * h) * kDim + 16 + l16] = o1[r] * inv;
  }
}

// ---------------------------------------------------------------------- host
extern "C" void kernel_launch(void* const* d_in, const int* in_sizes, int n_in,
                              void* d_out, int out_size, void* d_ws,
                              size_t ws_size, hipStream_t stream) {
  const float* x  = (const float*)d_in[0];
  const float* wq = (const float*)d_in[1];
  const float* wk = (const float*)d_in[2];
  const float* wv = (const float*)d_in[3];
  float* out = (float*)d_out;

  // workspace layout (bf16): x | w(q|k|v fused) | q | k | v^T  (~23 MB)
  char* ws = (char*)d_ws;
  __bf16* xb  = (__bf16*)ws;
  __bf16* wb  = xb + (size_t)kTok * kDim;
  __bf16* qb  = wb + (size_t)kNcol * kDim;
  __bf16* kb  = qb + (size_t)kTok * kDim;
  __bf16* vtb = kb + (size_t)kTok * kKvd;

  cvt_bf16_kernel<<<1024, 256, 0, stream>>>(x, xb, (kTok * kDim) / 8);
  cvt_bf16_kernel<<<256, 256, 0, stream>>>(wq, wb, (kDim * kDim) / 8);
  cvt_bf16_kernel<<<64, 256, 0, stream>>>(wk, wb + (size_t)kDim * kDim,
                                          (kKvd * kDim) / 8);
  cvt_bf16_kernel<<<64, 256, 0, stream>>>(
      wv, wb + (size_t)(kDim + kKvd) * kDim, (kKvd * kDim) / 8);

  qkv_proj_kernel<<<dim3(kTok / 64, kNcol / 256), 256, 0, stream>>>(
      xb, wb, qb, kb, vtb);

  gqa_attn_kernel<<<dim3(kSeq / 16, kGrp, kBs), 128, 0, stream>>>(qb, kb, vtb,
                                                                  out);
}
